// GNNModel_45268955300377
// MI455X (gfx1250) — matrix operand order repaired and checked by reading, not compiled, not run on hardware
//
#include <hip/hip_runtime.h>
#include <hip/hip_bf16.h>

#define NN 50000      // nodes (3125 * 16)
#define NE 500000     // edges
#define DIM 128       // D_IN == HID
#define NG 256        // graphs

typedef float v2f __attribute__((ext_vector_type(2)));
typedef float v8f __attribute__((ext_vector_type(8)));

#if defined(__has_builtin)
#  if __has_builtin(__builtin_amdgcn_wmma_f32_16x16x4_f32)
#    define USE_F32_WMMA 1
#  endif
#endif

#ifndef USE_F32_WMMA
typedef __bf16 v16bf __attribute__((ext_vector_type(16)));
#endif

// ---------------- zero fill ----------------
__global__ void k_zero(float* __restrict__ p, int n) {
    int i = blockIdx.x * blockDim.x + threadIdx.x;
    int stride = gridDim.x * blockDim.x;
    for (; i < n; i += stride) p[i] = 0.0f;
}

// ------------- edge scatter-add: agg[dst] += feat[src] -------------
// one wave32 per edge; each lane moves a float4 (32*16B = full 512B row)
__global__ __launch_bounds__(256) void k_scatter(const float* __restrict__ feat,
                                                 const int* __restrict__ src,
                                                 const int* __restrict__ dst,
                                                 float* __restrict__ agg) {
    const int lane = threadIdx.x & 31;
    const int e = blockIdx.x * 8 + (threadIdx.x >> 5);
    if (e >= NE) return;
    const int s = src[e];
    const int d = dst[e];
    float4 v = ((const float4*)(feat + (size_t)s * DIM))[lane];
    float* o = agg + (size_t)d * DIM + lane * 4;
    atomicAdd(o + 0, v.x);
    atomicAdd(o + 1, v.y);
    atomicAdd(o + 2, v.z);
    atomicAdd(o + 3, v.w);
}

// ------------- fused dual GEMM + bias + ReLU -------------
// out[n,o] = relu( sum_k A1[n,k]*W1[o,k] + sum_k A2[n,k]*W2[o,k] + bias[o] )
// One wave per 16x16 output tile. 25000 tiles, 8 waves/block -> 3125 blocks.
__global__ __launch_bounds__(256) void k_gemm_dual_relu(
    const float* __restrict__ A1, const float* __restrict__ W1,
    const float* __restrict__ A2, const float* __restrict__ W2,
    const float* __restrict__ bias, float* __restrict__ out) {
    const int lane = threadIdx.x & 31;
    const int wave = threadIdx.x >> 5;
    const int tile = blockIdx.x * 8 + wave;   // 0 .. 24999
    const int nt = tile >> 3;                 // node tile   0 .. 3124
    const int ot = tile & 7;                  // out-feature tile 0 .. 7
    const int m = lane & 15;                  // matrix row index within tile
    const int half = lane >> 4;

    const float* a1p = A1 + (size_t)(nt * 16 + m) * DIM;   // A row (node)
    const float* a2p = A2 + (size_t)(nt * 16 + m) * DIM;
    const float* w1p = W1 + (size_t)(ot * 16 + m) * DIM;   // B col n==m -> W row
    const float* w2p = W2 + (size_t)(ot * 16 + m) * DIM;

    v8f c = {};

#ifdef USE_F32_WMMA
    // f32 16x16x4: A frag elem r = A[m][k + 2*half + r]; B frag elem r = W[n][k + 2*half + r]
    // -> both are a contiguous f32 pair at offset k + 2*half (single b64 load each).
    for (int k = 0; k < DIM; k += 4) {
        const int ko = k + 2 * half;
        v2f a = *(const v2f*)(a1p + ko);
        v2f b = *(const v2f*)(w1p + ko);
        c = __builtin_amdgcn_wmma_f32_16x16x4_f32(false, a, false, b,
                                                  (short)0, c, false, false);
        v2f a2 = *(const v2f*)(a2p + ko);
        v2f b2 = *(const v2f*)(w2p + ko);
        c = __builtin_amdgcn_wmma_f32_16x16x4_f32(false, a2, false, b2,
                                                  (short)0, c, false, false);
    }
#else
    // bf16 16x16x32 fallback (codegen-confirmed builtin)
    for (int k = 0; k < DIM; k += 32) {
        v16bf a, b, a2, b2;
        for (int i = 0; i < 8; ++i) {
            a[i]      = (__bf16)a1p[k + 16 * half + i];
            a[i + 8]  = (__bf16)a1p[k + 16 * half + 8 + i];
            a2[i]     = (__bf16)a2p[k + 16 * half + i];
            a2[i + 8] = (__bf16)a2p[k + 16 * half + 8 + i];
        }
        for (int i = 0; i < 16; ++i) {
            b[i]  = (__bf16)w1p[k + 16 * half + i];
            b2[i] = (__bf16)w2p[k + 16 * half + i];
        }
        c = __builtin_amdgcn_wmma_f32_16x16x32_bf16(false, a, false, b,
                                                    (short)0, c, false, false);
        c = __builtin_amdgcn_wmma_f32_16x16x32_bf16(false, a2, false, b2,
                                                    (short)0, c, false, false);
    }
#endif

    const float bv = bias[ot * 16 + m];
    for (int r = 0; r < 8; ++r) {
        const int mm = r + 8 * half;                  // C/D: VGPR r -> rows r, r+8
        float v = c[r] + bv;
        out[(size_t)(nt * 16 + mm) * DIM + ot * 16 + m] = v > 0.0f ? v : 0.0f;
    }
}

// ------------- pool: per-node dot with W_out, scatter into per-graph sums -------------
__global__ __launch_bounds__(256) void k_pool(const float* __restrict__ h,
                                              const int* __restrict__ batch,
                                              const float* __restrict__ wout,
                                              float* __restrict__ gsum,
                                              float* __restrict__ gcnt) {
    const int lane = threadIdx.x & 31;
    const int n = blockIdx.x * 8 + (threadIdx.x >> 5);
    if (n >= NN) return;
    float4 v = ((const float4*)(h + (size_t)n * DIM))[lane];
    float4 w = ((const float4*)wout)[lane];
    float s = v.x * w.x + v.y * w.y + v.z * w.z + v.w * w.w;
    for (int off = 16; off > 0; off >>= 1) s += __shfl_down(s, off, 32);
    if (lane == 0) {
        const int g = batch[n];
        atomicAdd(&gsum[g], s);
        atomicAdd(&gcnt[g], 1.0f);
    }
}

__global__ void k_final(const float* __restrict__ gsum, const float* __restrict__ gcnt,
                        const float* __restrict__ bout, float* __restrict__ out) {
    const int g = threadIdx.x;
    if (g < NG) {
        float cnt = gcnt[g];
        cnt = cnt > 1.0f ? cnt : 1.0f;
        out[g] = gsum[g] / cnt + bout[0];
    }
}

extern "C" void kernel_launch(void* const* d_in, const int* in_sizes, int n_in,
                              void* d_out, int out_size, void* d_ws, size_t ws_size,
                              hipStream_t stream) {
    const float* x      = (const float*)d_in[0];
    const int*   eidx   = (const int*)d_in[1];   // [2, NE]: row0 = src, row1 = dst
    const int*   batch  = (const int*)d_in[2];
    const float* W1rel  = (const float*)d_in[3];
    const float* W1root = (const float*)d_in[4];
    const float* b1     = (const float*)d_in[5];
    const float* W2rel  = (const float*)d_in[6];
    const float* W2root = (const float*)d_in[7];
    const float* b2     = (const float*)d_in[8];
    const float* Wout   = (const float*)d_in[9];
    const float* bout   = (const float*)d_in[10];
    float* out = (float*)d_out;

    float* agg  = (float*)d_ws;                   // reused for both layers
    float* h1   = agg  + (size_t)NN * DIM;
    float* h2   = h1   + (size_t)NN * DIM;
    float* gsum = h2   + (size_t)NN * DIM;
    float* gcnt = gsum + NG;

    const int* src = eidx;
    const int* dst = eidx + NE;

    const int GEMM_BLOCKS = (NN / 16) * (DIM / 16) / 8;   // 3125
    const int SCAT_BLOCKS = NE / 8;                       // 62500
    const int POOL_BLOCKS = NN / 8;                       // 6250

    // layer 1
    k_zero<<<1024, 256, 0, stream>>>(agg, NN * DIM);
    k_zero<<<1, 256, 0, stream>>>(gsum, 2 * NG);          // gsum + gcnt contiguous
    k_scatter<<<SCAT_BLOCKS, 256, 0, stream>>>(x, src, dst, agg);
    k_gemm_dual_relu<<<GEMM_BLOCKS, 256, 0, stream>>>(agg, W1rel, x, W1root, b1, h1);

    // layer 2
    k_zero<<<1024, 256, 0, stream>>>(agg, NN * DIM);
    k_scatter<<<SCAT_BLOCKS, 256, 0, stream>>>(h1, src, dst, agg);
    k_gemm_dual_relu<<<GEMM_BLOCKS, 256, 0, stream>>>(agg, W2rel, h1, W2root, b2, h2);

    // mean-pool per graph fused with W_out projection
    k_pool<<<POOL_BLOCKS, 256, 0, stream>>>(h2, batch, Wout, gsum, gcnt);
    k_final<<<1, 256, 0, stream>>>(gsum, gcnt, bout, out);
}
